// GaussianModel_17480516895255
// MI455X (gfx1250) — compile-verified, matrix operand order checked by reading, not run on hardware
//
#include <hip/hip_runtime.h>
#include <hip/hip_bf16.h>
#include <math.h>

typedef __attribute__((ext_vector_type(16))) _Float16 v16h;
typedef __attribute__((ext_vector_type(8)))  float    v8f;

#define NUM_BINS    512
#define HALF_BIN    0.005f                 // BIN_RES / 2
#define CHUNK       512                    // gaussians per block
#define BLOCK       1024                   // 32 waves (wave32)
#define INV_SQRT2PI 0.3989422804014327f    // sqrt(0.5/pi)
#define LOG2E       1.4426950408889634f

__global__ __launch_bounds__(NUM_BINS) void hist_zero(float* out, int n) {
    int i = blockIdx.x * blockDim.x + threadIdx.x;
    if (i < n) out[i] = 0.0f;
}

__global__ __launch_bounds__(BLOCK) void gauss_hist(
    const float* __restrict__ means,     // N x 3
    const float* __restrict__ scan,      // 3
    const float* __restrict__ colours,   // N x 1
    const float* __restrict__ coeffs,    // N x 1
    const float* __restrict__ opac,      // N x 1
    const float* __restrict__ pas,       // N x 1
    float* __restrict__ out, int N)
{
    // Per-gaussian folded params: p = {r0, q, c1, c2}, cap = intensity, where
    //   q  = -0.5 * inv_sigma^2 * log2(e)   (exponent scale, pre-folded for v_exp_f32)
    //   contribution(g, j) = clamp((c1 + c2*d) * exp2(q*d*d), 0, cap), d = r_j - r0
    __shared__ float4 s_p[CHUNK];
    __shared__ float  s_cap[CHUNK];

    const int tid = threadIdx.x;
    const int gbase = blockIdx.x * CHUNK;

    if (tid < CHUNK) {
        int g = gbase + tid;
        float4 p = make_float4(0.f, 0.f, 0.f, 0.f);
        float cap = 0.f;
        if (g < N) {
            float mx = means[3 * g + 0] - scan[0];
            float my = means[3 * g + 1] - scan[1];
            float mz = means[3 * g + 2] - scan[2];
            float r0 = sqrtf(mx * mx + my * my + mz * mz);
            float col = colours[g];
            float inten = (1.0f / (1.0f + __expf(-opac[g]))) * col * col;   // sigmoid * colour^2
            float cf    = 1.0f / (1.0f + __expf(-coeffs[g]));               // sigmoid(coeff)
            float sig   = fmaxf(__expf(pas[g]), HALF_BIN);
            float inv   = 1.0f / sig;
            float s     = HALF_BIN * inten;                                  // fold pr-scale * intensity
            p.x = r0;
            p.y = -0.5f * inv * inv * LOG2E;          // exponent scale for exp2
            p.z = s * cf * INV_SQRT2PI * inv;         // coeff * pdf1 weight
            p.w = s * (1.0f - cf) * inv * inv;        // (1-coeff) * pdf2 weight
            cap = inten;                               // clip(pr,0,1)*inten == clamp(v,0,inten)
        }
        s_p[tid]   = p;
        s_cap[tid] = cap;
    }
    __syncthreads();

    const int lane = tid & 31;
    const int wave = tid >> 5;            // 0..31 -> bin tile
    const int col  = lane & 15;           // column within 16-bin tile
    const int half = lane >> 4;           // which K-half of the 32x16 B tile
    const int bin  = (wave << 4) + col;   // 0..511
    const float r  = HALF_BIN * (float)(bin + 1);
    const float inv_r2 = 1.0f / (r * r);  // DECAY=2 folded; hoisted out of epilogue

    float acc[16];
#pragma unroll
    for (int e = 0; e < 16; ++e) acc[e] = 0.0f;

    const int kbase = half * 16;
    for (int it = 0; it < CHUNK; it += 32) {
        const int kb = it + kbase;
#pragma unroll
        for (int e = 0; e < 16; ++e) {
            float4 p  = s_p[kb + e];                 // ds_load_b128, broadcast per half-wave
            float cap = s_cap[kb + e];
            float d   = r - p.x;
            float g   = __builtin_amdgcn_exp2f(d * d * p.y);   // v_exp_f32
            float v   = fmaf(p.w, d, p.z) * g;
            v = fminf(fmaxf(v, 0.0f), cap);          // v_maxmin_num_f32
            acc[e] += v;
        }
    }

    // Cross-lane / cross-slot reduction on the matrix unit:
    // B (32x16 f16): lane holds column `col`, K rows half*16+e. A = ones =>
    // D[m][n] = sum_k B[k][n] = histogram partial for bin wave*16+n (all rows equal).
    v16h a, b;
#pragma unroll
    for (int e = 0; e < 16; ++e) {
        a[e] = (_Float16)1.0f;
        b[e] = (_Float16)acc[e];
    }
    v8f c = {};
    c = __builtin_amdgcn_wmma_f32_16x16x32_f16(
        /*neg_a=*/false, a, /*neg_b=*/false, b,
        /*c_mod=*/(short)0, c, /*reuse_a=*/false, /*reuse_b=*/false);

    // c[0], lanes 0-15: D[0][lane] = partial for bin wave*16+lane.
    if (lane < 16) {
        atomicAdd(&out[bin], c[0] * inv_r2);
    }
}

extern "C" void kernel_launch(void* const* d_in, const int* in_sizes, int n_in,
                              void* d_out, int out_size, void* d_ws, size_t ws_size,
                              hipStream_t stream) {
    const float* means   = (const float*)d_in[0];
    const float* scan    = (const float*)d_in[1];
    const float* colours = (const float*)d_in[2];
    const float* coeffs  = (const float*)d_in[3];
    const float* opac    = (const float*)d_in[4];
    const float* pas     = (const float*)d_in[5];
    // d_in[6] = view_id; opacities has a single column, so it is always column 0.
    float* out = (float*)d_out;

    const int N = in_sizes[0] / 3;

    hist_zero<<<(out_size + NUM_BINS - 1) / NUM_BINS, NUM_BINS, 0, stream>>>(out, out_size);

    const int blocks = (N + CHUNK - 1) / CHUNK;   // 256 blocks for N=131072
    gauss_hist<<<blocks, BLOCK, 0, stream>>>(means, scan, colours, coeffs, opac, pas, out, N);
}
